// SpatialTransformer_24575802868328
// MI455X (gfx1250) — matrix operand order repaired and checked by reading, not compiled, not run on hardware
//
#include <hip/hip_runtime.h>
#include <hip/hip_bf16.h>
#include <math.h>

// ---------------------------------------------------------------------------
// CDNA5 (gfx1250) wave32 WMMA types/helpers
// ---------------------------------------------------------------------------
typedef __attribute__((ext_vector_type(16))) _Float16 v16h;
typedef __attribute__((ext_vector_type(8)))  _Float16 v8h;
typedef __attribute__((ext_vector_type(8)))  float    v8f;

#define WMMA_F16(a, b, c) \
  __builtin_amdgcn_wmma_f32_16x16x32_f16(false, (a), false, (b), (short)0, (c), false, false)

__device__ inline v16h cat8(v8h lo, v8h hi) {
  return __builtin_shufflevector(lo, hi, 0, 1, 2, 3, 4, 5, 6, 7,
                                 8, 9, 10, 11, 12, 13, 14, 15);
}

// Async 16B global -> LDS copy (gfx1250 GLOBAL_LOAD_ASYNC_TO_LDS_B128,
// tracked by ASYNCcnt; per-lane LDS byte address = low 32 bits of the
// generic shared pointer).
__device__ inline void async_copy16(void* lptr, const _Float16* g) {
  const unsigned int lds_off = (unsigned int)(uintptr_t)lptr;
  asm volatile("global_load_async_to_lds_b128 %0, %1, off"
               :: "v"(lds_off), "v"((unsigned long long)(uintptr_t)g)
               : "memory");
}
__device__ inline void wait_async() {
  asm volatile("s_wait_asynccnt 0" ::: "memory");
}

// A fragment (16x32 f16) from a row-major source, ld in halves.
// ISA: lane<16 -> M=lane, K=0..7 & 16..23; lane>=16 -> M=lane-16, K=8..15 &
// 24..31.  Per lane: two contiguous 16B runs.
__device__ inline v16h frag_a_rm(const _Float16* __restrict__ p, int ld) {
  const int lane = threadIdx.x & 31;
  const int row  = lane & 15;
  const int kb   = (lane & 16) ? 8 : 0;
  const _Float16* r = p + (size_t)row * ld + kb;
  return cat8(*(const v8h*)r, *(const v8h*)(r + 16));
}

// B fragment (32x16 f16) from a K-CONTIGUOUS source: p + n*ld + k.
// ISA: lane<16 -> N=lane, K=0..7 & 16..23; lane>=16 -> K=8..15 & 24..31.
// Per lane: two contiguous 16B runs.  nvalid guards columns.
__device__ inline v16h frag_b_kc(const _Float16* __restrict__ p, int ld, int nvalid) {
  const int lane = threadIdx.x & 31;
  const int n    = lane & 15;
  const int kb   = (lane & 16) ? 8 : 0;
  if (n < nvalid) {
    const _Float16* r = p + (size_t)n * ld + kb;
    return cat8(*(const v8h*)r, *(const v8h*)(r + 16));
  }
  v16h z = {};
  return z;
}

// ---------------------------------------------------------------------------
// WMMA GEMM: C[M,N] = A[M,K](f16,row-major) x BT[N,K](f16,pre-transposed)
// 256 thr = 8 waves; 64x128 tile; BK=64; wave (wm,wn) owns 16x64 (4 accs).
// Double-buffered LDS filled by async global->LDS DMA; OOB A rows clamped
// (tile row r only feeds output row r, which is masked at the store).
// ---------------------------------------------------------------------------
__global__ __launch_bounds__(256)
void gemm_f16(const _Float16* __restrict__ A, const _Float16* __restrict__ BT,
              const float* __restrict__ bias, const float* __restrict__ residual,
              float* __restrict__ Cf32, _Float16* __restrict__ Cf16,
              int M, int N, int K) {
  __shared__ __align__(16) _Float16 As[2][64][72];    // 64 x 64 (+8 pad)
  __shared__ __align__(16) _Float16 Bs[2][128][72];   // [n][k] 128 x 64 (+8 pad)

  const int tid  = threadIdx.x;
  const int wave = tid >> 5;
  const int wm   = wave & 3;           // 16-row strip
  const int wn   = wave >> 2;          // 64-col strip (0..1)
  const int bm   = blockIdx.y * 64;
  const int bn   = blockIdx.x * 128;

  const int ra  = tid >> 2;            // A staging: 4 thr/row, 16 halves each
  const int ca  = (tid & 3) * 16;
  const int rbs = tid >> 1;            // B staging: 2 thr/row, 32 halves each
  const int cb  = (tid & 1) * 32;

  const int gma = (bm + ra < M) ? (bm + ra) : (M - 1);
  const _Float16* gA = A + (size_t)gma * K;
  const _Float16* gB = BT + (size_t)(bn + rbs) * K;

  auto stage = [&](int k0, int buf) {
#pragma unroll
    for (int j = 0; j < 2; ++j)
      async_copy16(&As[buf][ra][ca + j * 8], gA + k0 + ca + j * 8);
#pragma unroll
    for (int j = 0; j < 4; ++j)
      async_copy16(&Bs[buf][rbs][cb + j * 8], gB + k0 + cb + j * 8);
  };

  v8f acc[4] = {};

  stage(0, 0);
  wait_async();
  __syncthreads();

  int buf = 0;
  for (int k0 = 0; k0 < K; k0 += 64, buf ^= 1) {
    if (k0 + 64 < K) stage(k0 + 64, buf ^ 1);   // overlap DMA with compute

#pragma unroll
    for (int ks = 0; ks < 2; ++ks) {
      const v16h af = frag_a_rm(&As[buf][wm * 16][ks * 32], 72);
      v16h bf[4];
#pragma unroll
      for (int g = 0; g < 4; ++g)
        bf[g] = frag_b_kc(&Bs[buf][wn * 64 + g * 16][ks * 32], 72, 16);
#pragma unroll
      for (int g = 0; g < 4; ++g)
        acc[g] = WMMA_F16(af, bf[g], acc[g]);
    }

    wait_async();
    __syncthreads();
  }

  const int lane = tid & 31;
  const int n    = lane & 15;
  const int rb   = (lane & 16) ? 8 : 0;
#pragma unroll
  for (int g = 0; g < 4; ++g) {
    const int gc = bn + wn * 64 + g * 16 + n;
    const float bval = bias ? bias[gc] : 0.f;
#pragma unroll
    for (int v = 0; v < 8; ++v) {
      const int gr = bm + wm * 16 + rb + v;
      if (gr < M) {
        float val = acc[g][v] + bval;
        const size_t idx = (size_t)gr * N + gc;
        if (residual) val += residual[idx];
        if (Cf32) Cf32[idx] = val;
        if (Cf16) Cf16[idx] = (_Float16)val;
      }
    }
  }
}

// ---------------------------------------------------------------------------
// Flash attention: per (b, h, 64 q rows); 128 thr = 4 waves, 16 q rows/wave.
// Q,K: [B*rows,1024] row-major (dh contiguous -> K-contiguous B frags).
// Vt : [B,1024,Lpad] (kv contiguous, zero padded) -> K-contiguous B frags.
// V-fragment loads are issued before the softmax LDS round-trip to hide
// global latency behind the exp/max VALU work.
// ---------------------------------------------------------------------------
__global__ __launch_bounds__(128)
void attention(const _Float16* __restrict__ Q, const _Float16* __restrict__ Km,
               const _Float16* __restrict__ Vt, _Float16* __restrict__ O,
               int Nq, int Lkv, int Lpad, float scale) {
  const int b    = blockIdx.z;
  const int h    = blockIdx.y;
  const int q0   = blockIdx.x * 64;
  const int wave = threadIdx.x >> 5;
  const int lane = threadIdx.x & 31;
  const int qr   = q0 + wave * 16;

  const _Float16* qptr  = Q  + ((size_t)(b * Nq + qr)) * 1024 + h * 64;
  const _Float16* kbase = Km + ((size_t)b * Lkv) * 1024 + h * 64;
  const _Float16* vtb   = Vt + ((size_t)b * 1024 + h * 64) * Lpad;

  __shared__ __align__(16) float    sS[4][16][36];
  __shared__ __align__(16) _Float16 sP[4][16][40];
  __shared__ float sM[4][16], sL[4][16], sA[4][16];

  const v16h qa0 = frag_a_rm(qptr, 1024);        // dh 0..31
  const v16h qa1 = frag_a_rm(qptr + 32, 1024);   // dh 32..63

  v8f o0 = {}, o1 = {}, o2 = {}, o3 = {};
  if (lane < 16) { sM[wave][lane] = -1e30f; sL[wave][lane] = 0.f; }
  __syncthreads();

  const int n  = lane & 15;
  const int rb = (lane & 16) ? 8 : 0;

  for (int kv0 = 0; kv0 < Lkv; kv0 += 32) {
    const _Float16* kc0 = kbase + (size_t)kv0 * 1024;
    const _Float16* kc1 = kc0 + (size_t)16 * 1024;
    const int nv0 = Lkv - kv0, nv1 = nv0 - 16;

    // --- K fragments first, then S = scale * Q K^T ---
    const v16h kf00 = frag_b_kc(kc0,      1024, nv0);
    const v16h kf01 = frag_b_kc(kc0 + 32, 1024, nv0);
    const v16h kf10 = frag_b_kc(kc1,      1024, nv1);
    const v16h kf11 = frag_b_kc(kc1 + 32, 1024, nv1);
    v8f s0 = {}, s1 = {};
    s0 = WMMA_F16(qa0, kf00, s0);
    s0 = WMMA_F16(qa1, kf01, s0);
    s1 = WMMA_F16(qa0, kf10, s1);
    s1 = WMMA_F16(qa1, kf11, s1);

    // --- issue V fragment loads early (independent of softmax) ---
    const v16h vf0 = frag_b_kc(vtb + kv0,                     Lpad, 16);
    const v16h vf1 = frag_b_kc(vtb + (size_t)16 * Lpad + kv0, Lpad, 16);
    const v16h vf2 = frag_b_kc(vtb + (size_t)32 * Lpad + kv0, Lpad, 16);
    const v16h vf3 = frag_b_kc(vtb + (size_t)48 * Lpad + kv0, Lpad, 16);

#pragma unroll
    for (int v = 0; v < 8; ++v) {
      float x0 = s0[v] * scale, x1 = s1[v] * scale;
      if (kv0 + n      >= Lkv) x0 = -1e30f;
      if (kv0 + 16 + n >= Lkv) x1 = -1e30f;
      sS[wave][rb + v][n]      = x0;
      sS[wave][rb + v][16 + n] = x1;
    }
    __syncthreads();

    // --- online softmax: lane r<16 owns row r ---
    if (lane < 16) {
      float cmax = -1e30f;
#pragma unroll
      for (int c = 0; c < 32; ++c) cmax = fmaxf(cmax, sS[wave][lane][c]);
      const float mold  = sM[wave][lane];
      const float mnew  = fmaxf(mold, cmax);
      const float alpha = __expf(mold - mnew);
      float psum = 0.f;
#pragma unroll
      for (int c = 0; c < 32; ++c) {
        const float p = __expf(sS[wave][lane][c] - mnew);
        sP[wave][lane][c] = (_Float16)p;
        psum += p;
      }
      sM[wave][lane] = mnew;
      sL[wave][lane] = sL[wave][lane] * alpha + psum;
      sA[wave][lane] = alpha;
    }
    __syncthreads();

    // --- rescale O, accumulate P x V ---
    const v16h pf = frag_a_rm(&sP[wave][0][0], 40);
#pragma unroll
    for (int v = 0; v < 8; ++v) {
      const float al = sA[wave][rb + v];
      o0[v] *= al; o1[v] *= al; o2[v] *= al; o3[v] *= al;
    }
    o0 = WMMA_F16(pf, vf0, o0);
    o1 = WMMA_F16(pf, vf1, o1);
    o2 = WMMA_F16(pf, vf2, o2);
    o3 = WMMA_F16(pf, vf3, o3);
    __syncthreads();
  }

  // --- normalize, store ---
#pragma unroll
  for (int g = 0; g < 4; ++g) {
    const v8f acc = (g == 0) ? o0 : (g == 1) ? o1 : (g == 2) ? o2 : o3;
#pragma unroll
    for (int v = 0; v < 8; ++v) {
      const int row = rb + v;
      const float val = acc[v] / sL[wave][row];
      O[((size_t)(b * Nq + qr + row)) * 1024 + h * 64 + g * 16 + n] = (_Float16)val;
    }
  }
}

// ---------------------------------------------------------------------------
// Normalization / pointwise / layout kernels
// ---------------------------------------------------------------------------
__device__ inline float block_reduce(float v, float* buf) {
  const int tid = threadIdx.x;
  buf[tid] = v;
  __syncthreads();
  for (int s = blockDim.x >> 1; s > 0; s >>= 1) {
    if (tid < s) buf[tid] += buf[tid + s];
    __syncthreads();
  }
  const float r = buf[0];
  __syncthreads();
  return r;
}

__global__ __launch_bounds__(256)
void groupnorm_nhwc(const float* __restrict__ x, const float* __restrict__ s,
                    const float* __restrict__ bia, _Float16* __restrict__ out) {
  __shared__ float buf[256];
  const int b = blockIdx.x >> 5, g = blockIdx.x & 31;
  const float* xp = x + ((size_t)b * 1024 + g * 32) * 1600;
  float sum = 0.f, sq = 0.f;
  for (int i = threadIdx.x; i < 32 * 1600; i += 256) {
    const float v = xp[i]; sum += v; sq += v * v;
  }
  sum = block_reduce(sum, buf);
  sq  = block_reduce(sq, buf);
  const float mean = sum * (1.f / 51200.f);
  const float rstd = rsqrtf(sq * (1.f / 51200.f) - mean * mean + 1e-6f);
  for (int i = threadIdx.x; i < 32 * 1600; i += 256) {
    const int c = g * 32 + i / 1600, p = i % 1600;
    const float v = (xp[i] - mean) * rstd * s[c] + bia[c];
    out[((size_t)b * 1600 + p) * 1024 + c] = (_Float16)v;
  }
}

__global__ __launch_bounds__(256)
void layernorm(const float* __restrict__ x, const float* __restrict__ s,
               const float* __restrict__ bia, _Float16* __restrict__ out) {
  __shared__ float buf[256];
  const float* xp = x + (size_t)blockIdx.x * 1024;
  float sum = 0.f, sq = 0.f;
  for (int i = threadIdx.x; i < 1024; i += 256) {
    const float v = xp[i]; sum += v; sq += v * v;
  }
  sum = block_reduce(sum, buf);
  sq  = block_reduce(sq, buf);
  const float mean = sum * (1.f / 1024.f);
  const float rstd = rsqrtf(sq * (1.f / 1024.f) - mean * mean + 1e-5f);
  for (int i = threadIdx.x; i < 1024; i += 256)
    out[(size_t)blockIdx.x * 1024 + i] = (_Float16)((xp[i] - mean) * rstd * s[i] + bia[i]);
}

__global__ void convert_f32_f16(const float* __restrict__ s, _Float16* __restrict__ d, int n) {
  const int i = blockIdx.x * 256 + threadIdx.x;
  if (i < n) d[i] = (_Float16)s[i];
}

// f32 [K,N] row-major -> f16 [N,K] row-major (weights pre-transpose).
__global__ __launch_bounds__(256)
void convert_transpose(const float* __restrict__ src, _Float16* __restrict__ dst,
                       int K, int N) {
  __shared__ float tile[32][33];
  const int k0 = blockIdx.y * 32, n0 = blockIdx.x * 32;
  const int tx = threadIdx.x & 31, ty = threadIdx.x >> 5;
  for (int j = ty; j < 32; j += 8)
    tile[j][tx] = src[(size_t)(k0 + j) * N + n0 + tx];
  __syncthreads();
  for (int j = ty; j < 32; j += 8)
    dst[(size_t)(n0 + j) * K + k0 + tx] = (_Float16)tile[tx][j];
}

// V [B*L,1024] f16 -> Vt [B,1024,Lpad] f16, zero-padding kv >= L.
__global__ __launch_bounds__(256)
void transpose_v(const _Float16* __restrict__ v, _Float16* __restrict__ vt,
                 int L, int Lpad) {
  __shared__ _Float16 tile[32][33];
  const int b = blockIdx.z;
  const int l0 = blockIdx.y * 32, d0 = blockIdx.x * 32;
  const int tx = threadIdx.x & 31, ty = threadIdx.x >> 5;
  for (int j = ty; j < 32; j += 8) {
    const int l = l0 + j;
    tile[j][tx] = (l < L) ? v[((size_t)(b * L + l)) * 1024 + d0 + tx] : (_Float16)0.f;
  }
  __syncthreads();
  for (int j = ty; j < 32; j += 8)
    vt[((size_t)b * 1024 + d0 + j) * Lpad + l0 + tx] = tile[tx][j];
}

__global__ void geglu(const _Float16* __restrict__ h, _Float16* __restrict__ g, int rows) {
  const int i = blockIdx.x * 256 + threadIdx.x;
  if (i >= rows * 4096) return;
  const int r = i >> 12, c = i & 4095;
  const float a  = (float)h[(size_t)r * 8192 + c];
  const float x  = (float)h[(size_t)r * 8192 + 4096 + c];
  const float ge = 0.5f * x * (1.f + erff(x * 0.70710678118654752f));
  g[i] = (_Float16)(a * ge);
}

__global__ void nhwc_to_nchw_add(const float* __restrict__ y, const float* __restrict__ xin,
                                 float* __restrict__ out, int total) {
  const int i = blockIdx.x * 256 + threadIdx.x;
  if (i >= total) return;
  const int p = i % 1600, c = (i / 1600) & 1023, b = i / (1600 * 1024);
  out[i] = y[((size_t)(b * 1600 + p)) * 1024 + c] + xin[i];
}

// ---------------------------------------------------------------------------
// Host orchestration
// ---------------------------------------------------------------------------
extern "C" void kernel_launch(void* const* d_in, const int* in_sizes, int n_in,
                              void* d_out, int out_size, void* d_ws, size_t ws_size,
                              hipStream_t stream) {
  const float* x     = (const float*)d_in[0];
  const float* ctx   = (const float*)d_in[1];
  const float* gn_s  = (const float*)d_in[2];
  const float* gn_b  = (const float*)d_in[3];
  const float* pin_w = (const float*)d_in[4];
  const float* pin_b = (const float*)d_in[5];
  const float* ln1_s = (const float*)d_in[6];
  const float* ln1_b = (const float*)d_in[7];
  const float* wq1   = (const float*)d_in[8];
  const float* wk1   = (const float*)d_in[9];
  const float* wv1   = (const float*)d_in[10];
  const float* wo1   = (const float*)d_in[11];
  const float* bo1   = (const float*)d_in[12];
  const float* ln2_s = (const float*)d_in[13];
  const float* ln2_b = (const float*)d_in[14];
  const float* wq2   = (const float*)d_in[15];
  const float* wk2   = (const float*)d_in[16];
  const float* wv2   = (const float*)d_in[17];
  const float* wo2   = (const float*)d_in[18];
  const float* bo2   = (const float*)d_in[19];
  const float* ln3_s = (const float*)d_in[20];
  const float* ln3_b = (const float*)d_in[21];
  const float* ffw1  = (const float*)d_in[22];
  const float* ffb1  = (const float*)d_in[23];
  const float* ffw2  = (const float*)d_in[24];
  const float* ffb2  = (const float*)d_in[25];
  const float* poutw = (const float*)d_in[26];
  const float* poutb = (const float*)d_in[27];
  float* out = (float*)d_out;

  const int Btok = 3200;
  const int Lctx = 2 * 77;

  char* wp = (char*)d_ws;
  auto alloc = [&](size_t bytes) -> void* {
    void* p = (void*)wp;
    wp += (bytes + 255) & ~(size_t)255;
    return p;
  };
  const size_t MH = (size_t)1024 * 1024;

  _Float16* w_pin  = (_Float16*)alloc(2 * MH);        // all weights stored [N,K]
  _Float16* w_q1   = (_Float16*)alloc(2 * MH);
  _Float16* w_k1   = (_Float16*)alloc(2 * MH);
  _Float16* w_v1   = (_Float16*)alloc(2 * MH);
  _Float16* w_o1   = (_Float16*)alloc(2 * MH);
  _Float16* w_q2   = (_Float16*)alloc(2 * MH);
  _Float16* w_k2   = (_Float16*)alloc(2 * MH);
  _Float16* w_v2   = (_Float16*)alloc(2 * MH);
  _Float16* w_o2   = (_Float16*)alloc(2 * MH);
  _Float16* w_ff1  = (_Float16*)alloc(2 * 8 * MH);
  _Float16* w_ff2  = (_Float16*)alloc(2 * 4 * MH);
  _Float16* w_pout = (_Float16*)alloc(2 * MH);
  _Float16* ctx_h  = (_Float16*)alloc(2ull * Lctx * 1024);
  float*    t      = (float*)   alloc(4ull * Btok * 1024);
  _Float16* t_h    = (_Float16*)alloc(2ull * Btok * 1024);
  _Float16* hx     = (_Float16*)alloc(2ull * Btok * 1024);
  _Float16* a_ln   = (_Float16*)alloc(2ull * Btok * 1024);
  _Float16* qb     = (_Float16*)alloc(2ull * Btok * 1024);
  _Float16* kbuf   = (_Float16*)alloc(2ull * Btok * 1024);
  _Float16* vbuf   = (_Float16*)alloc(2ull * Btok * 1024);
  _Float16* vts    = (_Float16*)alloc(2ull * 2 * 1024 * 1600);  // self Vt
  _Float16* k2b    = (_Float16*)alloc(2ull * Lctx * 1024);
  _Float16* v2b    = (_Float16*)alloc(2ull * Lctx * 1024);
  _Float16* vtc    = (_Float16*)alloc(2ull * 2 * 1024 * 96);    // cross Vt (padded)
  _Float16* attn   = (_Float16*)alloc(2ull * Btok * 1024);
  _Float16* ffh    = (_Float16*)alloc(2ull * Btok * 8192);
  _Float16* ffg    = (_Float16*)alloc(2ull * Btok * 4096);
  float*    yb     = (float*)   alloc(4ull * Btok * 1024);

  auto cvtT = [&](const float* s, _Float16* d, int K, int N) {
    convert_transpose<<<dim3(N / 32, K / 32), 256, 0, stream>>>(s, d, K, N);
  };
  auto gemm = [&](const _Float16* A, const _Float16* BT, const float* bias,
                  const float* res, float* Cf, _Float16* Ch, int M, int Nn, int K) {
    dim3 grid(Nn / 128, (M + 63) / 64);
    gemm_f16<<<grid, 256, 0, stream>>>(A, BT, bias, res, Cf, Ch, M, Nn, K);
  };

  // Weight pre-transpose+convert; context convert.
  cvtT(pin_w, w_pin, 1024, 1024);
  cvtT(wq1, w_q1, 1024, 1024);  cvtT(wk1, w_k1, 1024, 1024);
  cvtT(wv1, w_v1, 1024, 1024);  cvtT(wo1, w_o1, 1024, 1024);
  cvtT(wq2, w_q2, 1024, 1024);  cvtT(wk2, w_k2, 1024, 1024);
  cvtT(wv2, w_v2, 1024, 1024);  cvtT(wo2, w_o2, 1024, 1024);
  cvtT(ffw1, w_ff1, 1024, 8192);
  cvtT(ffw2, w_ff2, 4096, 1024);
  cvtT(poutw, w_pout, 1024, 1024);
  convert_f32_f16<<<(Lctx * 1024 + 255) / 256, 256, 0, stream>>>(ctx, ctx_h, Lctx * 1024);

  // GroupNorm + tokens, proj_in
  groupnorm_nhwc<<<64, 256, 0, stream>>>(x, gn_s, gn_b, hx);
  gemm(hx, w_pin, pin_b, nullptr, t, nullptr, Btok, 1024, 1024);

  // Self-attention block
  layernorm<<<Btok, 256, 0, stream>>>(t, ln1_s, ln1_b, a_ln);
  gemm(a_ln, w_q1, nullptr, nullptr, nullptr, qb, Btok, 1024, 1024);
  gemm(a_ln, w_k1, nullptr, nullptr, nullptr, kbuf, Btok, 1024, 1024);
  gemm(a_ln, w_v1, nullptr, nullptr, nullptr, vbuf, Btok, 1024, 1024);
  transpose_v<<<dim3(32, 50, 2), 256, 0, stream>>>(vbuf, vts, 1600, 1600);
  attention<<<dim3(25, 16, 2), 128, 0, stream>>>(qb, kbuf, vts, attn, 1600, 1600, 1600, 0.125f);
  gemm(attn, w_o1, bo1, t, t, nullptr, Btok, 1024, 1024);

  // Cross-attention block
  layernorm<<<Btok, 256, 0, stream>>>(t, ln2_s, ln2_b, a_ln);
  gemm(a_ln, w_q2, nullptr, nullptr, nullptr, qb, Btok, 1024, 1024);
  gemm(ctx_h, w_k2, nullptr, nullptr, nullptr, k2b, Lctx, 1024, 1024);
  gemm(ctx_h, w_v2, nullptr, nullptr, nullptr, v2b, Lctx, 1024, 1024);
  transpose_v<<<dim3(32, 3, 2), 256, 0, stream>>>(v2b, vtc, 77, 96);
  attention<<<dim3(25, 16, 2), 128, 0, stream>>>(qb, k2b, vtc, attn, 1600, 77, 96, 0.125f);
  gemm(attn, w_o2, bo2, t, t, nullptr, Btok, 1024, 1024);

  // GEGLU feed-forward
  layernorm<<<Btok, 256, 0, stream>>>(t, ln3_s, ln3_b, a_ln);
  gemm(a_ln, w_ff1, ffb1, nullptr, nullptr, ffh, Btok, 8192, 1024);
  geglu<<<(Btok * 4096 + 255) / 256, 256, 0, stream>>>(ffh, ffg, Btok);
  gemm(ffg, w_ff2, ffb2, t, t, t_h, Btok, 1024, 4096);

  // proj_out + residual back to NCHW
  gemm(t_h, w_pout, poutb, nullptr, yb, nullptr, Btok, 1024, 1024);
  const int total = 2 * 1024 * 1600;
  nhwc_to_nchw_add<<<(total + 255) / 256, 256, 0, stream>>>(yb, x, out, total);
}
